// GVQEncoder_30743375905291
// MI455X (gfx1250) — matrix operand order repaired and checked by reading, not compiled, not run on hardware
//
#include <hip/hip_runtime.h>
#include <hip/hip_bf16.h>

// ---------------------------------------------------------------------------
// GVQ graph encoder for gfx1250: SAGEConv x3 (+BN+ReLU) -> tanh linear head
// -> mu/sigma heads -> batchnorm.  All GEMMs via v_wmma_f32_16x16x32_bf16.
// ---------------------------------------------------------------------------

typedef __attribute__((ext_vector_type(16))) __bf16 v16bf;
typedef __attribute__((ext_vector_type(8)))  float  v8f;

#define N_NODES 65536
#define N_EDGES 1048576
#define C_IN    128
#define C_H     128
#define C_OUT   64
#define NB      256
#define LIN_INK 16384
#define LIN_OUTN 6400

static __device__ __forceinline__ void atomic_add_f32(float* p, float v) {
  __hip_atomic_fetch_add(p, v, __ATOMIC_RELAXED, __HIP_MEMORY_SCOPE_AGENT);
}

// --------------------------- scatter-mean helpers --------------------------

__global__ void k_count(const int* __restrict__ dst, float* __restrict__ cnt, int E) {
  int e = blockIdx.x * blockDim.x + threadIdx.x;
  if (e < E) atomic_add_f32(&cnt[dst[e]], 1.0f);
}

__global__ void k_scatter(const float* __restrict__ x, const int* __restrict__ src,
                          const int* __restrict__ dst, float* __restrict__ agg,
                          int E, int C) {
  long long t = (long long)blockIdx.x * blockDim.x + threadIdx.x;
  int per = C >> 2;                       // float4 groups per edge
  long long total = (long long)E * per;
  if (t >= total) return;
  int e  = (int)(t / per);
  int c4 = (int)(t % per) << 2;
  const float4 v = *(const float4*)(x + (size_t)src[e] * C + c4);
  float* a = agg + (size_t)dst[e] * C + c4;
  atomic_add_f32(a + 0, v.x);
  atomic_add_f32(a + 1, v.y);
  atomic_add_f32(a + 2, v.z);
  atomic_add_f32(a + 3, v.w);
}

__global__ void k_mean(const float* __restrict__ agg, const float* __restrict__ cnt,
                       float* __restrict__ mean, int Nn, int C) {
  size_t i = (size_t)blockIdx.x * blockDim.x + threadIdx.x;
  if (i >= (size_t)Nn * C) return;
  int n = (int)(i / C);
  mean[i] = agg[i] / fmaxf(cnt[n], 1.0f);
}

// ------------------------------ WMMA GEMM ----------------------------------
// out[M,N] (+)= act( A[M,K] @ W[N,K]^T + bias )
// Block = 8 waves (2 M-groups x 4 N-cols), block tile 64(M) x 64(N).
// Each wave owns a 32x16 output (two 16x16 WMMA tiles sharing one B frag).
// fp32 operands loaded as float4 and converted to bf16 in-register.
// act: 0 = none, 1 = tanh.  accum: 1 = C initialized from out (no bias/act).

static __device__ __forceinline__ void pack8(v16bf& v, int base, float4 x, float4 y) {
  v[base + 0] = (__bf16)x.x; v[base + 1] = (__bf16)x.y;
  v[base + 2] = (__bf16)x.z; v[base + 3] = (__bf16)x.w;
  v[base + 4] = (__bf16)y.x; v[base + 5] = (__bf16)y.y;
  v[base + 6] = (__bf16)y.z; v[base + 7] = (__bf16)y.w;
}

__global__ void __launch_bounds__(256)
k_gemm_wmma(const float* __restrict__ A, const float* __restrict__ W,
            const float* __restrict__ bias, float* __restrict__ out,
            int M, int N, int K, int accum, int act) {
  const int lane = threadIdx.x & 31;
  const int wid  = threadIdx.x >> 5;
  const int wm   = wid >> 2;            // 0..1  -> 32 rows each
  const int wn   = wid & 3;             // 0..3  -> 16 cols each
  const int m0   = blockIdx.y * 64 + wm * 32;
  const int n0   = blockIdx.x * 64 + wn * 16;

  const int r  = lane & 15;             // row (A) / col (B) within 16-tile
  const int hi = lane >> 4;             // lane-half selector

  // C/D layout: lane -> col = lane&15 ; VGPR i -> row = i + 8*(lane>>4)
  const int ccol = r;
  const int crow = hi << 3;

  v8f c0, c1;
  if (accum) {
#pragma unroll
    for (int i = 0; i < 8; ++i) {
      c0[i] = out[(size_t)(m0 + crow + i) * N + n0 + ccol];
      c1[i] = out[(size_t)(m0 + 16 + crow + i) * N + n0 + ccol];
    }
  } else {
#pragma unroll
    for (int i = 0; i < 8; ++i) { c0[i] = 0.0f; c1[i] = 0.0f; }
  }

  // A fragment (16x32 bf16): lane half hi needs floats kc+hi*8+[0..7] and
  // kc+16+hi*8+[0..7].  With base advanced by hi*8 floats (= 2 float4s):
  // float4 offsets kc/4+{0,1} and kc/4+{4,5}.
  const float4* Ap0 = (const float4*)(A + (size_t)(m0 + r) * K) + hi * 2;
  const float4* Ap1 = (const float4*)(A + (size_t)(m0 + 16 + r) * K) + hi * 2;
  // B fragment (32x16 bf16): lane half hi needs floats kc+hi*8+[0..7] and
  // kc+16+hi*8+[0..7]: base advanced by hi*8 floats: offsets kc/4+{0,1,4,5}.
  const float4* Wp  = (const float4*)(W + (size_t)(n0 + r) * K) + hi * 2;

  for (int k0 = 0; k0 < K; k0 += 64) {
#pragma unroll
    for (int ch = 0; ch < 2; ++ch) {
      const int kq = (k0 >> 2) + ch * 8;      // float4 index of 32-chunk base
      float4 a0a = Ap0[kq],     a0b = Ap0[kq + 1];
      float4 a0c = Ap0[kq + 4], a0d = Ap0[kq + 5];
      float4 a1a = Ap1[kq],     a1b = Ap1[kq + 1];
      float4 a1c = Ap1[kq + 4], a1d = Ap1[kq + 5];
      float4 b0  = Wp[kq],      b1  = Wp[kq + 1];
      float4 b2  = Wp[kq + 4],  b3  = Wp[kq + 5];

      v16bf a0, a1, b;
      pack8(a0, 0, a0a, a0b); pack8(a0, 8, a0c, a0d);
      pack8(a1, 0, a1a, a1b); pack8(a1, 8, a1c, a1d);
      pack8(b, 0, b0, b1);    pack8(b, 8, b2, b3);

      c0 = __builtin_amdgcn_wmma_f32_16x16x32_bf16(
              false, a0, false, b, (short)0, c0, false, false);
      c1 = __builtin_amdgcn_wmma_f32_16x16x32_bf16(
              false, a1, false, b, (short)0, c1, false, false);
    }
  }

#pragma unroll
  for (int i = 0; i < 8; ++i) {
    float v0 = c0[i];
    float v1 = c1[i];
    if (bias) {
      float bb = bias[n0 + ccol];
      v0 += bb;
      v1 += bb;
    }
    if (act == 1) { v0 = tanhf(v0); v1 = tanhf(v1); }
    out[(size_t)(m0 + crow + i) * N + n0 + ccol] = v0;
    out[(size_t)(m0 + 16 + crow + i) * N + n0 + ccol] = v1;
  }
}

// --------------------------- batchnorm (training) --------------------------

__global__ void k_bnstats(const float* __restrict__ x, int M, int C, int rowsPer,
                          float* __restrict__ sum, float* __restrict__ sumsq) {
  int cch = blockIdx.x * blockDim.x + threadIdx.x;
  if (cch >= C) return;
  int r0 = blockIdx.y * rowsPer;
  int r1 = r0 + rowsPer;
  if (r1 > M) r1 = M;
  float s = 0.0f, s2 = 0.0f;
  for (int rr = r0; rr < r1; ++rr) {
    float v = x[(size_t)rr * C + cch];
    s += v;
    s2 += v * v;
  }
  atomic_add_f32(&sum[cch], s);
  atomic_add_f32(&sumsq[cch], s2);
}

__global__ void k_bnapply(const float* __restrict__ x, const float* __restrict__ sum,
                          const float* __restrict__ sumsq, const float* __restrict__ gamma,
                          const float* __restrict__ beta, float* __restrict__ y,
                          int M, int C, int relu) {
  size_t i = (size_t)blockIdx.x * blockDim.x + threadIdx.x;
  if (i >= (size_t)M * C) return;
  int cch = (int)(i % C);
  float invM = 1.0f / (float)M;
  float m = sum[cch] * invM;
  float v = sumsq[cch] * invM - m * m;
  float rs = rsqrtf(v + 1e-5f);
  float o = (x[i] - m) * rs * gamma[cch] + beta[cch];
  if (relu) o = fmaxf(o, 0.0f);
  y[i] = o;
}

// ------------------------------- launcher ----------------------------------

extern "C" void kernel_launch(void* const* d_in, const int* in_sizes, int n_in,
                              void* d_out, int out_size, void* d_ws, size_t ws_size,
                              hipStream_t stream) {
  (void)in_sizes; (void)n_in; (void)out_size; (void)ws_size;

  const float* x    = (const float*)d_in[0];
  const int*   ei   = (const int*)d_in[1];
  // d_in[2] = batch: layout is exactly repeat(arange(B), MAXN) -> pure reshape.
  const float* W1l = (const float*)d_in[3];
  const float* b1l = (const float*)d_in[4];
  const float* W1r = (const float*)d_in[5];
  const float* g1  = (const float*)d_in[6];
  const float* be1 = (const float*)d_in[7];
  const float* W2l = (const float*)d_in[8];
  const float* b2l = (const float*)d_in[9];
  const float* W2r = (const float*)d_in[10];
  const float* g2  = (const float*)d_in[11];
  const float* be2 = (const float*)d_in[12];
  const float* W3l = (const float*)d_in[13];
  const float* b3l = (const float*)d_in[14];
  const float* W3r = (const float*)d_in[15];
  const float* g3  = (const float*)d_in[16];
  const float* be3 = (const float*)d_in[17];
  const float* Wlin = (const float*)d_in[18];
  const float* blin = (const float*)d_in[19];
  const float* Wmu  = (const float*)d_in[20];
  const float* bmu  = (const float*)d_in[21];
  const float* Wsig = (const float*)d_in[22];
  const float* bsig = (const float*)d_in[23];
  const float* gfx  = (const float*)d_in[24];
  const float* bfx  = (const float*)d_in[25];
  const float* gfs  = (const float*)d_in[26];
  const float* bfs  = (const float*)d_in[27];

  const int* src = ei;
  const int* dst = ei + N_EDGES;

  float* ws = (float*)d_ws;
  size_t off = 0;
  float* cnt   = ws + off; off += N_NODES;
  float* agg   = ws + off; off += (size_t)N_NODES * C_H;
  float* meanb = ws + off; off += (size_t)N_NODES * C_H;
  float* hpre  = ws + off; off += (size_t)N_NODES * C_H;
  float* h1    = ws + off; off += (size_t)N_NODES * C_H;
  float* h2    = ws + off; off += (size_t)N_NODES * C_H;
  float* h3    = ws + off; off += (size_t)N_NODES * C_OUT;
  float* htanh = ws + off; off += (size_t)NB * LIN_OUTN;
  float* mupre = ws + off; off += (size_t)NB * LIN_OUTN;
  float* sgpre = ws + off; off += (size_t)NB * LIN_OUTN;
  float* bsum  = ws + off; off += LIN_OUTN;
  float* bsq   = ws + off; off += LIN_OUTN;

  auto zero = [&](float* p, size_t n) {
    hipMemsetAsync(p, 0, n * sizeof(float), stream);
  };
  auto gemm = [&](const float* A, const float* W, const float* bias, float* o,
                  int M, int Nc, int K, int accum, int act) {
    dim3 g((unsigned)((Nc + 63) / 64), (unsigned)(M / 64));
    k_gemm_wmma<<<g, 256, 0, stream>>>(A, W, bias, o, M, Nc, K, accum, act);
  };
  auto bn = [&](const float* in, const float* gamma, const float* beta, float* o,
                int M, int C, int relu) {
    zero(bsum, C);
    zero(bsq, C);
    int rows = (M < 1024) ? M : 1024;
    dim3 gs((unsigned)((C + 127) / 128), (unsigned)((M + rows - 1) / rows));
    k_bnstats<<<gs, 128, 0, stream>>>(in, M, C, rows, bsum, bsq);
    size_t tot = (size_t)M * C;
    k_bnapply<<<(unsigned)((tot + 255) / 256), 256, 0, stream>>>(
        in, bsum, bsq, gamma, beta, o, M, C, relu);
  };
  auto sage = [&](const float* xin, int Cin, const float* Wl, const float* bl,
                  const float* Wr, int Cout, float* hp) {
    zero(agg, (size_t)N_NODES * Cin);
    long long tot = (long long)N_EDGES * (Cin >> 2);
    k_scatter<<<(unsigned)((tot + 255) / 256), 256, 0, stream>>>(xin, src, dst, agg,
                                                                N_EDGES, Cin);
    size_t nm = (size_t)N_NODES * Cin;
    k_mean<<<(unsigned)((nm + 255) / 256), 256, 0, stream>>>(agg, cnt, meanb,
                                                             N_NODES, Cin);
    gemm(meanb, Wl, bl, hp, N_NODES, Cout, Cin, /*accum=*/0, /*act=*/0);
    gemm(xin,   Wr, nullptr, hp, N_NODES, Cout, Cin, /*accum=*/1, /*act=*/0);
  };

  // In-degree counts (shared by all three layers)
  zero(cnt, N_NODES);
  k_count<<<(N_EDGES + 255) / 256, 256, 0, stream>>>(dst, cnt, N_EDGES);

  // Layer 1: x(128) -> h1(128)
  sage(x, C_IN, W1l, b1l, W1r, C_H, hpre);
  bn(hpre, g1, be1, h1, N_NODES, C_H, /*relu=*/1);
  // Layer 2: h1(128) -> h2(128)
  sage(h1, C_H, W2l, b2l, W2r, C_H, hpre);
  bn(hpre, g2, be2, h2, N_NODES, C_H, 1);
  // Layer 3: h2(128) -> h3(64)
  sage(h2, C_H, W3l, b3l, W3r, C_OUT, hpre);
  bn(hpre, g3, be3, h3, N_NODES, C_OUT, 1);

  // h3 [65536,64] row-major == [256,16384] row-major: tanh linear head
  gemm(h3, Wlin, blin, htanh, NB, LIN_OUTN, LIN_INK, 0, /*tanh=*/1);
  // mu / sigma heads
  gemm(htanh, Wmu,  bmu,  mupre, NB, LIN_OUTN, LIN_OUTN, 0, 0);
  gemm(htanh, Wsig, bsig, sgpre, NB, LIN_OUTN, LIN_OUTN, 0, 0);

  // Final batchnorms straight into d_out (mu first, then sigma)
  float* outp = (float*)d_out;
  bn(mupre, gfx, bfx, outp, NB, LIN_OUTN, 0);
  bn(sgpre, gfs, bfs, outp + (size_t)NB * LIN_OUTN, NB, LIN_OUTN, 0);
}
